// QKVAttention_16475494547544
// MI455X (gfx1250) — compile-verified, matrix operand order checked by reading, not run on hardware
//
#include <hip/hip_runtime.h>

#define C_DIM 768
#define SEQ   2048
#define BATCH 4
#define HEADS 16
#define HDIM  48
#define DPAD  64
#define SCALE 0.14433756729740643f  // 48^-0.5

typedef __attribute__((ext_vector_type(16))) __bf16 bf16x16;
typedef __attribute__((ext_vector_type(8)))  __bf16 bf16x8;
typedef __attribute__((ext_vector_type(4)))  __bf16 bf16x4;
typedef __attribute__((ext_vector_type(8)))  float  f32x8;

// D = A(16x32 bf16) * B(32x16 bf16) + C(16x16 f32), f32 accumulate.
__device__ __forceinline__ f32x8 wmma_bf16(bf16x16 a, bf16x16 b, f32x8 c) {
  return __builtin_amdgcn_wmma_f32_16x16x32_bf16(false, a, false, b, (short)0, c,
                                                 false, false);
}

// A/B fragment loader. ISA 16-bit operand layout: lane L (0-15) holds row
// (L&15) with K = kb..kb+7 in elems 0..7 and K = kb+16..kb+23 in elems 8..15,
// where kb = (lane>=16)*8. Caller passes p = &tile[row][kb]; we grab two
// 16-byte chunks (k and k+16) -> two b128 loads.
__device__ __forceinline__ bf16x16 load_frag(const __bf16* p) {
  bf16x8 lo = *(const bf16x8*)(p);
  bf16x8 hi = *(const bf16x8*)(p + 16);
  bf16x16 r;
#pragma unroll
  for (int i = 0; i < 8; ++i) { r[i] = lo[i]; r[8 + i] = hi[i]; }
  return r;
}

// Async global->LDS copy, 16 bytes per lane (GVS mode: SGPR base + VGPR byte
// offset).  Tracked on ASYNCcnt; no VGPR data round-trip.
__device__ __forceinline__ void async_copy_b128(uint32_t lds_byte_off,
                                                const void* sbase,
                                                uint32_t gbyte_off) {
  asm volatile("global_load_async_to_lds_b128 %0, %1, %2"
               :: "v"(lds_byte_off), "v"(gbyte_off), "s"(sbase)
               : "memory");
}
__device__ __forceinline__ void wait_asynccnt0() {
  asm volatile("s_wait_asynccnt 0x0" ::: "memory");
}

// ---------------------------------------------------------------------------
// Kernel 1: fused QKV projection.  out[m, c] = sum_k x[m,k] * W[c,k] + b[c]
// 128x128 tile per workgroup, 8 waves (2M x 4N), K-step 32, bf16 LDS tiles.
// Q -> [B,H,N,64] bf16 pre-scaled; K -> [B,H,N,64] bf16; V -> Vt [B,H,64,N].
// (fp32 sources must be converted to bf16, so staging uses the VALU path.)
// ---------------------------------------------------------------------------
__global__ __launch_bounds__(256) void qkv_gemm_kernel(
    const float* __restrict__ x,
    const float* __restrict__ wq, const float* __restrict__ bq,
    const float* __restrict__ wk, const float* __restrict__ bk,
    const float* __restrict__ wv, const float* __restrict__ bv,
    __bf16* __restrict__ Qb, __bf16* __restrict__ Kb, __bf16* __restrict__ Vt)
{
  __shared__ __bf16 As[128][40];   // pitch 40 (80B): conflict-free b128 reads
  __shared__ __bf16 Bs[128][40];

  const int rowBase = blockIdx.x * 128;
  const int mat     = blockIdx.y / 6;          // 0=Q 1=K 2=V
  const int colBase = (blockIdx.y % 6) * 128;

  const float* __restrict__ W    = (mat == 0) ? wq : (mat == 1) ? wk : wv;
  const float* __restrict__ bias = (mat == 0) ? bq : (mat == 1) ? bk : bv;

  const int tid   = threadIdx.x;
  const int wid   = tid >> 5;
  const int lane  = tid & 31;
  const int waveM = wid >> 2;        // 0..1 -> 64-row block
  const int waveN = wid & 3;         // 0..3 -> 32-col block
  const int fr    = lane & 15;
  const int kb    = (lane >> 4) * 8;
  const int lhalf = lane >> 4;

  const int lr = tid >> 3;           // staging: 0..31
  const int lc = (tid & 7) * 4;      // staging: 0,4,...,28

  f32x8 acc[4][2] = {};

  for (int k0 = 0; k0 < C_DIM; k0 += 32) {
    __syncthreads();
#pragma unroll
    for (int rr = 0; rr < 4; ++rr) {
      const int r = lr + rr * 32;
      float4 va = *(const float4*)(x + (size_t)(rowBase + r) * C_DIM + k0 + lc);
      bf16x4 pa = {(__bf16)va.x, (__bf16)va.y, (__bf16)va.z, (__bf16)va.w};
      *(bf16x4*)(&As[r][lc]) = pa;
      float4 vb = *(const float4*)(W + (size_t)(colBase + r) * C_DIM + k0 + lc);
      bf16x4 pb = {(__bf16)vb.x, (__bf16)vb.y, (__bf16)vb.z, (__bf16)vb.w};
      *(bf16x4*)(&Bs[r][lc]) = pb;
    }
    __syncthreads();

    bf16x16 afr[4];
#pragma unroll
    for (int mi = 0; mi < 4; ++mi)
      afr[mi] = load_frag(&As[waveM * 64 + mi * 16 + fr][kb]);
#pragma unroll
    for (int ni = 0; ni < 2; ++ni) {
      bf16x16 bfr = load_frag(&Bs[waveN * 32 + ni * 16 + fr][kb]);
#pragma unroll
      for (int mi = 0; mi < 4; ++mi)
        acc[mi][ni] = wmma_bf16(afr[mi], bfr, acc[mi][ni]);
    }
  }

  // C layout: element e of v8f in lane l -> row e + 8*(l>=16), col (l&15).
#pragma unroll
  for (int mi = 0; mi < 4; ++mi) {
#pragma unroll
    for (int ni = 0; ni < 2; ++ni) {
#pragma unroll
      for (int e = 0; e < 8; ++e) {
        const int m = rowBase + waveM * 64 + mi * 16 + e + 8 * lhalf;
        const int c = colBase + waveN * 32 + ni * 16 + fr;
        const float v = acc[mi][ni][e] + bias[c];
        const int b    = m >> 11;          // / 2048
        const int nseq = m & 2047;
        const int h    = c / HDIM;
        const int d    = c - h * HDIM;
        const size_t bh = (size_t)b * HEADS + h;
        if (mat == 0)
          Qb[(bh * SEQ + nseq) * DPAD + d] = (__bf16)(v * SCALE);
        else if (mat == 1)
          Kb[(bh * SEQ + nseq) * DPAD + d] = (__bf16)v;
        else
          Vt[(bh * DPAD + d) * SEQ + nseq] = (__bf16)v;  // transposed store
      }
    }
  }
}

// ---------------------------------------------------------------------------
// Kernel 2: flash attention.  grid = (qtile, b*H).  8 waves x 16 query rows.
// K/V tiles are staged cooperatively into LDS once per 64-key step with
// GLOBAL_LOAD_ASYNC_TO_LDS_B128 (bf16->bf16 byte copy, ASYNCcnt tracked),
// removing the 8x redundant per-wave global reads.  Online softmax; the
// P tile round-trips per-wave LDS for the C->A operand relayout.
// ---------------------------------------------------------------------------
__global__ __launch_bounds__(256) void attention_kernel(
    const __bf16* __restrict__ Qb, const __bf16* __restrict__ Kb,
    const __bf16* __restrict__ Vt, const float* __restrict__ x,
    float* __restrict__ y32)
{
  __shared__ __bf16 Ks[64][72];        // keys: row = key, col = d (padded 64)
  __shared__ __bf16 Vs[64][72];        // V^T : row = d (48 used), col = key
  __shared__ __bf16 Plds[8][16][72];   // per-wave P tile

  const int qtile = blockIdx.x;
  const int bh    = blockIdx.y;
  const int tid   = threadIdx.x;
  const int wid   = tid >> 5;
  const int lane  = tid & 31;
  const int fr    = lane & 15;
  const int kb    = (lane >> 4) * 8;
  const int lhalf = lane >> 4;

  const __bf16* __restrict__ Qp = Qb + (size_t)bh * SEQ * DPAD;
  const __bf16* __restrict__ Kp = Kb + (size_t)bh * SEQ * DPAD;
  const __bf16* __restrict__ Vp = Vt + (size_t)bh * DPAD * SEQ;

  const int q0 = qtile * 128 + wid * 16;

  // Q fragments are loop-invariant: D=64(padded) -> two K-steps of 32.
  bf16x16 aq0 = load_frag(Qp + (size_t)(q0 + fr) * DPAD + kb);
  bf16x16 aq1 = load_frag(Qp + (size_t)(q0 + fr) * DPAD + 32 + kb);

  // Cooperative staging: 64x64 bf16 tile = 16 elems (32B) per thread.
  const int sr = tid >> 2;          // 0..63 : tile row
  const int sc = (tid & 3) * 16;    // 0,16,32,48 : tile col chunk
  const uint32_t ldsK = (uint32_t)(uintptr_t)&Ks[sr][sc];  // addrspace(3) off
  const uint32_t ldsV = (uint32_t)(uintptr_t)&Vs[sr][sc];

  f32x8 yacc[3] = {};                  // 16 q-rows x 48 d (3 N-tiles)
  float rmax[8], rsum[8];
#pragma unroll
  for (int e = 0; e < 8; ++e) { rmax[e] = -3.0e38f; rsum[e] = 0.0f; }

  for (int mt = 0; mt < SEQ; mt += 64) {
    __syncthreads();                   // previous K/V tile fully consumed
    const uint32_t kOff = (uint32_t)(((mt + sr) * DPAD + sc) * 2);
    const uint32_t vOff = (uint32_t)((sr * SEQ + mt + sc) * 2);
    async_copy_b128(ldsK,      Kp, kOff);
    async_copy_b128(ldsK + 16, Kp, kOff + 16);
    async_copy_b128(ldsV,      Vp, vOff);
    async_copy_b128(ldsV + 16, Vp, vOff + 16);
    wait_asynccnt0();
    __syncthreads();                   // tile visible to all waves

    // S = Q * K^T for this 16x64 block (scale folded into Q).
    f32x8 s[4] = {};
#pragma unroll
    for (int sub = 0; sub < 4; ++sub) {
      const int kr = sub * 16 + fr;
      s[sub] = wmma_bf16(aq0, load_frag(&Ks[kr][kb]), s[sub]);
      s[sub] = wmma_bf16(aq1, load_frag(&Ks[kr][32 + kb]), s[sub]);
    }
    // Online softmax.  Row r of S lives at v8f element e=r%8, half lhalf=r/8;
    // columns live across the 16 lanes of each half -> xor-shuffle reduce.
#pragma unroll
    for (int e = 0; e < 8; ++e) {
      float m = fmaxf(fmaxf(s[0][e], s[1][e]), fmaxf(s[2][e], s[3][e]));
      m = fmaxf(m, __shfl_xor(m, 1));
      m = fmaxf(m, __shfl_xor(m, 2));
      m = fmaxf(m, __shfl_xor(m, 4));
      m = fmaxf(m, __shfl_xor(m, 8));
      const float nm = fmaxf(rmax[e], m);
      const float sc2 = __expf(rmax[e] - nm);
      rmax[e] = nm;
      float ps = 0.0f;
#pragma unroll
      for (int sub = 0; sub < 4; ++sub) {
        const float p = __expf(s[sub][e] - nm);
        ps += p;
        Plds[wid][e + 8 * lhalf][sub * 16 + fr] = (__bf16)p;  // C->A relayout
      }
      ps += __shfl_xor(ps, 1);
      ps += __shfl_xor(ps, 2);
      ps += __shfl_xor(ps, 4);
      ps += __shfl_xor(ps, 8);
      rsum[e] = rsum[e] * sc2 + ps;
      yacc[0][e] *= sc2; yacc[1][e] *= sc2; yacc[2][e] *= sc2;
    }
    // yacc += P(16x64) * V(64x48); per-wave LDS, same-wave DS ops are ordered.
#pragma unroll
    for (int ks = 0; ks < 2; ++ks) {
      bf16x16 ap = load_frag(&Plds[wid][fr][ks * 32 + kb]);
#pragma unroll
      for (int t = 0; t < 3; ++t)
        yacc[t] = wmma_bf16(ap, load_frag(&Vs[t * 16 + fr][ks * 32 + kb]),
                            yacc[t]);
    }
  }

  const int b = bh >> 4;
  const int h = bh & 15;
#pragma unroll
  for (int t = 0; t < 3; ++t) {
#pragma unroll
    for (int e = 0; e < 8; ++e) {
      const int nq = q0 + e + 8 * lhalf;
      const int c  = h * HDIM + t * 16 + fr;
      const size_t idx = ((size_t)b * SEQ + nq) * C_DIM + c;
      y32[idx] = yacc[t][e] / rsum[e] + x[idx];   // y = attn(x) + x, fp32
    }
  }
}

// ---------------------------------------------------------------------------
// Kernel 3: output projection.  out = y @ wo^T + bo + y + x  (all fp32 adds).
// ---------------------------------------------------------------------------
__global__ __launch_bounds__(256) void out_proj_kernel(
    const float* __restrict__ y32, const float* __restrict__ x,
    const float* __restrict__ wo, const float* __restrict__ bo,
    float* __restrict__ out)
{
  __shared__ __bf16 As[128][40];
  __shared__ __bf16 Bs[128][40];

  const int rowBase = blockIdx.x * 128;
  const int colBase = blockIdx.y * 128;

  const int tid   = threadIdx.x;
  const int wid   = tid >> 5;
  const int lane  = tid & 31;
  const int waveM = wid >> 2;
  const int waveN = wid & 3;
  const int fr    = lane & 15;
  const int kb    = (lane >> 4) * 8;
  const int lhalf = lane >> 4;
  const int lr    = tid >> 3;
  const int lc    = (tid & 7) * 4;

  f32x8 acc[4][2] = {};

  for (int k0 = 0; k0 < C_DIM; k0 += 32) {
    __syncthreads();
#pragma unroll
    for (int rr = 0; rr < 4; ++rr) {
      const int r = lr + rr * 32;
      float4 va = *(const float4*)(y32 + (size_t)(rowBase + r) * C_DIM + k0 + lc);
      bf16x4 pa = {(__bf16)va.x, (__bf16)va.y, (__bf16)va.z, (__bf16)va.w};
      *(bf16x4*)(&As[r][lc]) = pa;
      float4 vb = *(const float4*)(wo + (size_t)(colBase + r) * C_DIM + k0 + lc);
      bf16x4 pb = {(__bf16)vb.x, (__bf16)vb.y, (__bf16)vb.z, (__bf16)vb.w};
      *(bf16x4*)(&Bs[r][lc]) = pb;
    }
    __syncthreads();

    bf16x16 afr[4];
#pragma unroll
    for (int mi = 0; mi < 4; ++mi)
      afr[mi] = load_frag(&As[waveM * 64 + mi * 16 + fr][kb]);
#pragma unroll
    for (int ni = 0; ni < 2; ++ni) {
      bf16x16 bfr = load_frag(&Bs[waveN * 32 + ni * 16 + fr][kb]);
#pragma unroll
      for (int mi = 0; mi < 4; ++mi)
        acc[mi][ni] = wmma_bf16(afr[mi], bfr, acc[mi][ni]);
    }
  }

#pragma unroll
  for (int mi = 0; mi < 4; ++mi)
#pragma unroll
    for (int ni = 0; ni < 2; ++ni)
#pragma unroll
      for (int e = 0; e < 8; ++e) {
        const int m = rowBase + waveM * 64 + mi * 16 + e + 8 * lhalf;
        const int c = colBase + waveN * 32 + ni * 16 + fr;
        const size_t idx = (size_t)m * C_DIM + c;
        out[idx] = acc[mi][ni][e] + bo[c] + y32[idx] + x[idx];
      }
}

// ---------------------------------------------------------------------------
extern "C" void kernel_launch(void* const* d_in, const int* in_sizes, int n_in,
                              void* d_out, int out_size, void* d_ws, size_t ws_size,
                              hipStream_t stream) {
  (void)in_sizes; (void)n_in; (void)out_size; (void)ws_size;

  const float* x  = (const float*)d_in[0];
  const float* wq = (const float*)d_in[1];
  const float* bq = (const float*)d_in[2];
  const float* wk = (const float*)d_in[3];
  const float* bk = (const float*)d_in[4];
  const float* wv = (const float*)d_in[5];
  const float* bv = (const float*)d_in[6];
  const float* wo = (const float*)d_in[7];
  const float* bo = (const float*)d_in[8];

  char* ws = (char*)d_ws;
  const size_t qkvBytes = (size_t)BATCH * HEADS * SEQ * DPAD * sizeof(__bf16); // 16 MB
  __bf16* Qb  = (__bf16*)(ws);
  __bf16* Kb  = (__bf16*)(ws + qkvBytes);
  __bf16* Vt  = (__bf16*)(ws + 2 * qkvBytes);
  float*  y32 = (float*)(ws + 3 * qkvBytes);   // 25 MB

  // Zero Q and K so the D=48..63 pad contributes nothing to QK^T.
  hipMemsetAsync(Qb, 0, 2 * qkvBytes, stream);

  qkv_gemm_kernel<<<dim3(64, 18), 256, 0, stream>>>(x, wq, bq, wk, bk, wv, bv,
                                                    Qb, Kb, Vt);
  attention_kernel<<<dim3(16, 64), 256, 0, stream>>>(Qb, Kb, Vt, x, y32);
  out_proj_kernel<<<dim3(64, 6), 256, 0, stream>>>(y32, x, wo, bo, (float*)d_out);
}
